// Scaled_Dot_Product_Attention_27659589386888
// MI455X (gfx1250) — compile-verified, matrix operand order checked by reading, not run on hardware
//
#include <hip/hip_runtime.h>

typedef __attribute__((ext_vector_type(16))) _Float16 v16h;
typedef __attribute__((ext_vector_type(8)))  _Float16 v8h;
typedef __attribute__((ext_vector_type(4)))  _Float16 v4h;
typedef __attribute__((ext_vector_type(8)))  float    v8f;

// Problem constants (from reference): B=4, H=16, S=2048, DK=DV=64.
#define ATT_S   2048
#define ATT_D   64
#define ATT_BH  64
// 1/sqrt(64) * log2(e): scores land in the exp2 domain.
#define QSCALE  0.18033688011112042f

// f16 element count of one [BH][S][D] array and ws requirement (Qh, Kh, Vth).
#define ARR_ELT ((size_t)ATT_BH * ATT_S * ATT_D)          // 8,388,608
#define WS_NEED (3 * ARR_ELT * sizeof(_Float16))          // 50,331,648 B

// ---------------------------------------------------------------------------
// Common helpers
// ---------------------------------------------------------------------------

// exp2 that lowers to a single v_exp_f32.
__device__ __forceinline__ float fast_exp2(float x) {
#if __has_builtin(__builtin_amdgcn_exp2f)
  return __builtin_amdgcn_exp2f(x);
#else
  return exp2f(x);
#endif
}

// Two 16B reads stitched into one 32B WMMA operand fragment.
__device__ __forceinline__ v16h frag16(const _Float16* lo, const _Float16* hi) {
  v8h a = *(const v8h*)lo;
  v8h b = *(const v8h*)hi;
  return __builtin_shufflevector(a, b, 0, 1, 2, 3, 4, 5, 6, 7,
                                 8, 9, 10, 11, 12, 13, 14, 15);
}

// Async global->LDS 16B copy (CDNA5 VGLOBAL async op, tracked by ASYNCcnt).
__device__ __forceinline__ void async_copy16(unsigned lds_off, const _Float16* g) {
  unsigned long long ga = (unsigned long long)(uintptr_t)g;
  asm volatile("global_load_async_to_lds_b128 %0, %1, off"
               :: "v"(lds_off), "v"(ga) : "memory");
}
__device__ __forceinline__ void wait_async0() {
  asm volatile("s_wait_asynccnt 0" ::: "memory");
}

// Load 8 consecutive f32, scale, convert into frag[off..off+7] (fallback path).
__device__ __forceinline__ void load8_f16(const float* __restrict__ p, float scale,
                                          v16h& frag, int off) {
  float4 a = *(const float4*)(p);
  float4 b = *(const float4*)(p + 4);
  frag[off + 0] = (_Float16)(a.x * scale);
  frag[off + 1] = (_Float16)(a.y * scale);
  frag[off + 2] = (_Float16)(a.z * scale);
  frag[off + 3] = (_Float16)(a.w * scale);
  frag[off + 4] = (_Float16)(b.x * scale);
  frag[off + 5] = (_Float16)(b.y * scale);
  frag[off + 6] = (_Float16)(b.z * scale);
  frag[off + 7] = (_Float16)(b.w * scale);
}

// ---------------------------------------------------------------------------
// Pre-pass 1: Q (scaled) and K f32 -> f16, contiguous.
// ---------------------------------------------------------------------------
__global__ __launch_bounds__(256) void cvt_qk_kernel(
    const float* __restrict__ Q, const float* __restrict__ K,
    _Float16* __restrict__ Qh, _Float16* __restrict__ Kh) {
  const size_t nq8 = ARR_ELT / 8;
  size_t i = (size_t)blockIdx.x * 256 + threadIdx.x;   // one 8-elem chunk each
  const float* src;
  _Float16* dst;
  float scale;
  if (i < nq8) { src = Q; dst = Qh; scale = QSCALE; }
  else         { src = K; dst = Kh; scale = 1.0f; i -= nq8; }
  const float* p = src + i * 8;
  float4 a = *(const float4*)(p);
  float4 b = *(const float4*)(p + 4);
  v8h o;
  o[0] = (_Float16)(a.x * scale); o[1] = (_Float16)(a.y * scale);
  o[2] = (_Float16)(a.z * scale); o[3] = (_Float16)(a.w * scale);
  o[4] = (_Float16)(b.x * scale); o[5] = (_Float16)(b.y * scale);
  o[6] = (_Float16)(b.z * scale); o[7] = (_Float16)(b.w * scale);
  *(v8h*)(dst + i * 8) = o;
}

// ---------------------------------------------------------------------------
// Pre-pass 2: V f32 [bh][key][d] -> f16 transposed Vth [bh][d][key].
// Thread handles 2 d-values x 8 keys (coalesced float2 reads, 16B writes).
// ---------------------------------------------------------------------------
__global__ __launch_bounds__(256) void transpose_v_kernel(
    const float* __restrict__ V, _Float16* __restrict__ Vth) {
  const int dp  = threadIdx.x & 31;          // d pair index
  const int kgl = threadIdx.x >> 5;          // local key-group (0..7)
  const int bh  = blockIdx.x >> 5;           // head (64)
  const int kg  = (blockIdx.x & 31) * 8 + kgl;  // key-group (0..255)
  const int d0  = dp * 2;
  const int k0  = kg * 8;
  const float* src = V + ((size_t)bh * ATT_S + k0) * ATT_D + d0;
  v8h r0, r1;
#pragma unroll
  for (int j = 0; j < 8; ++j) {
    float2 x = *(const float2*)(src + (size_t)j * ATT_D);
    r0[j] = (_Float16)x.x;
    r1[j] = (_Float16)x.y;
  }
  _Float16* dst = Vth + ((size_t)bh * ATT_D + d0) * ATT_S + k0;
  *(v8h*)(dst)         = r0;
  *(v8h*)(dst + ATT_S) = r1;
}

// ---------------------------------------------------------------------------
// Main kernel (fast path): flash attention, causal, f16 WMMA + f32 accum.
// One WG = 128 query rows of one head (8 wave32, 16 rows each). K/V blocks
// (already f16, V pre-transposed) stream through double-buffered LDS via
// global_load_async_to_lds_b128 — no VGPR transit, no in-loop conversions.
// Scores computed TRANSPOSED (S^T = K*Q^T): softmax needs one xor-16
// shuffle; P^T repacks in-lane into the B operand of O^T = V^T * P^T.
// ---------------------------------------------------------------------------
struct __align__(16) KVBuf {
  _Float16 ks[32][72];   // K block row-major: [key][d], 144 B stride (skewed)
  _Float16 vt[64][40];   // V^T block:        [d][key],  80 B stride (skewed)
};

__device__ __forceinline__ void stage_async(KVBuf& b,
                                            const _Float16* __restrict__ Khb,
                                            const _Float16* __restrict__ Vthb,
                                            int kb, int tid) {
  // K: thread -> (key = tid>>3, 16B segment = tid&7).
  const int key = tid >> 3, seg = tid & 7;
  async_copy16((unsigned)(uintptr_t)&b.ks[key][seg * 8],
               Khb + (size_t)(kb + key) * ATT_D + seg * 8);
  // V^T: thread -> (d = tid>>2, 16B segment = tid&3).
  const int d = tid >> 2, sg = tid & 3;
  async_copy16((unsigned)(uintptr_t)&b.vt[d][sg * 8],
               Vthb + (size_t)d * ATT_S + kb + sg * 8);
}

__global__ __launch_bounds__(256) void fa_causal_wmma_async_kernel(
    const _Float16* __restrict__ Qh, const _Float16* __restrict__ Kh,
    const _Float16* __restrict__ Vth, float* __restrict__ O) {
  __shared__ KVBuf sbuf[2];

  const int tid  = threadIdx.x;
  const int lane = tid & 31;
  const int wave = tid >> 5;
  const int bh   = blockIdx.x >> 4;
  const int wq0  = (blockIdx.x & 15) << 7;     // WG query strip base
  const int q0   = wq0 + wave * 16;            // this wave's 16-row tile
  const int n    = lane & 15;
  const int h    = lane >> 4;

  const _Float16* __restrict__ Qhb  = Qh  + (size_t)bh * ATT_S * ATT_D;
  const _Float16* __restrict__ Khb  = Kh  + (size_t)bh * ATT_S * ATT_D;
  const _Float16* __restrict__ Vthb = Vth + (size_t)bh * ATT_D * ATT_S;
  float* __restrict__          Ob   = O   + (size_t)bh * ATT_S * ATT_D;

  // Q as B operand (scale already folded by pre-pass); pure b128 loads.
  v16h qf[2];
  {
    const _Float16* qrow = Qhb + (size_t)(q0 + n) * ATT_D;
#pragma unroll
    for (int c = 0; c < 2; ++c)
      qf[c] = frag16(qrow + 32 * c + 8 * h, qrow + 32 * c + 16 + 8 * h);
  }

  v8f acc[4];
#pragma unroll
  for (int c = 0; c < 4; ++c)
#pragma unroll
    for (int r = 0; r < 8; ++r) acc[c][r] = 0.0f;

  float m_run = -1e30f;
  float l_run = 0.0f;

  const int nb = (wq0 >> 5) + 4;

  stage_async(sbuf[0], Khb, Vthb, 0, tid);
  wait_async0();
  __syncthreads();

  for (int blk = 0; blk < nb; ++blk) {
    const int  kb        = blk << 5;
    const bool have_next = (blk + 1 < nb);
    if (have_next)                      // async copy for next block, hidden
      stage_async(sbuf[(blk + 1) & 1], Khb, Vthb, kb + 32, tid);

    if (kb < q0 + 16) {
      KVBuf& b = sbuf[blk & 1];

      // ---- scores^T: two 16-key tiles x two d-chunk WMMAs ----
      v8f st[2];
#pragma unroll
      for (int t = 0; t < 2; ++t) {
        const int row = 16 * t + n;
        v16h ka0 = frag16(&b.ks[row][8 * h],      &b.ks[row][16 + 8 * h]);
        v16h ka1 = frag16(&b.ks[row][32 + 8 * h], &b.ks[row][48 + 8 * h]);
        v8f c0;
#pragma unroll
        for (int r = 0; r < 8; ++r) c0[r] = 0.0f;
        c0 = __builtin_amdgcn_wmma_f32_16x16x32_f16(false, ka0, false, qf[0],
                                                    (short)0, c0, false, false);
        c0 = __builtin_amdgcn_wmma_f32_16x16x32_f16(false, ka1, false, qf[1],
                                                    (short)0, c0, false, false);
        st[t] = c0;
      }

      // ---- causal mask (diagonal-crossing blocks only) ----
      if (kb + 31 > q0) {
        const int q = q0 + n;
#pragma unroll
        for (int t = 0; t < 2; ++t)
#pragma unroll
          for (int r = 0; r < 8; ++r) {
            const int key = kb + 16 * t + 8 * h + r;
            if (key > q) st[t][r] = -1e30f;
          }
      }

      // ---- online softmax in exp2 domain (q spans lanes {n, n+16}) ----
      float bm = -1e30f;
#pragma unroll
      for (int t = 0; t < 2; ++t)
#pragma unroll
        for (int r = 0; r < 8; ++r) bm = fmaxf(bm, st[t][r]);
      bm = fmaxf(bm, __shfl_xor(bm, 16, 32));
      const float m_new = fmaxf(m_run, bm);
      const float corr  = fast_exp2(m_run - m_new);

      v16h pb;
      float rsum = 0.0f;
#pragma unroll
      for (int t = 0; t < 2; ++t)
#pragma unroll
        for (int r = 0; r < 8; ++r) {
          const float p = fast_exp2(st[t][r] - m_new);
          rsum += p;
          pb[8 * t + r] = (_Float16)p;
        }
      rsum += __shfl_xor(rsum, 16, 32);
      l_run = l_run * corr + rsum;
      m_run = m_new;

      // ---- O^T += V^T * P^T ----
#pragma unroll
      for (int c = 0; c < 4; ++c) {
#pragma unroll
        for (int r = 0; r < 8; ++r) acc[c][r] *= corr;
        const int vd = 16 * c + n;
        v16h va = frag16(&b.vt[vd][8 * h], &b.vt[vd][16 + 8 * h]);
        acc[c] = __builtin_amdgcn_wmma_f32_16x16x32_f16(false, va, false, pb,
                                                        (short)0, acc[c],
                                                        false, false);
      }
    }

    wait_async0();     // our async copies for next block done before barrier
    __syncthreads();
  }

  const float inv_l = 1.0f / l_run;
  float* orow = Ob + (size_t)(q0 + n) * ATT_D;
#pragma unroll
  for (int c = 0; c < 4; ++c) {
    float4 o0, o1;
    o0.x = acc[c][0] * inv_l; o0.y = acc[c][1] * inv_l;
    o0.z = acc[c][2] * inv_l; o0.w = acc[c][3] * inv_l;
    o1.x = acc[c][4] * inv_l; o1.y = acc[c][5] * inv_l;
    o1.z = acc[c][6] * inv_l; o1.w = acc[c][7] * inv_l;
    *(float4*)(orow + 16 * c + 8 * h)     = o0;
    *(float4*)(orow + 16 * c + 8 * h + 4) = o1;
  }
}

// ---------------------------------------------------------------------------
// Fallback (ws too small): self-contained kernel, converts in-loop (round-2).
// ---------------------------------------------------------------------------
struct StageRegs { float4 k0, k1; float2 v0, v1, v2, v3; };

__device__ __forceinline__ StageRegs stage_load(const float* __restrict__ Kb,
                                                const float* __restrict__ Vb,
                                                int kb, int tid) {
  StageRegs s;
  const int key = tid >> 3, dc = tid & 7;
  const float* ksrc = Kb + (size_t)(kb + key) * ATT_D + dc * 8;
  s.k0 = *(const float4*)(ksrc);
  s.k1 = *(const float4*)(ksrc + 4);
  const int d0 = (tid & 31) * 2, kh = tid >> 5;
  const float* vsrc = Vb + (size_t)(kb + kh * 4) * ATT_D + d0;
  s.v0 = *(const float2*)(vsrc);
  s.v1 = *(const float2*)(vsrc + ATT_D);
  s.v2 = *(const float2*)(vsrc + 2 * ATT_D);
  s.v3 = *(const float2*)(vsrc + 3 * ATT_D);
  return s;
}

__device__ __forceinline__ void stage_store(KVBuf& b, const StageRegs& s, int tid) {
  const int key = tid >> 3, dc = tid & 7;
  v8h kv;
  kv[0] = (_Float16)s.k0.x; kv[1] = (_Float16)s.k0.y;
  kv[2] = (_Float16)s.k0.z; kv[3] = (_Float16)s.k0.w;
  kv[4] = (_Float16)s.k1.x; kv[5] = (_Float16)s.k1.y;
  kv[6] = (_Float16)s.k1.z; kv[7] = (_Float16)s.k1.w;
  *(v8h*)&b.ks[key][dc * 8] = kv;
  const int d0 = (tid & 31) * 2, kh = tid >> 5;
  v4h a, c;
  a[0] = (_Float16)s.v0.x; a[1] = (_Float16)s.v1.x;
  a[2] = (_Float16)s.v2.x; a[3] = (_Float16)s.v3.x;
  c[0] = (_Float16)s.v0.y; c[1] = (_Float16)s.v1.y;
  c[2] = (_Float16)s.v2.y; c[3] = (_Float16)s.v3.y;
  *(v4h*)&b.vt[d0][kh * 4]     = a;
  *(v4h*)&b.vt[d0 + 1][kh * 4] = c;
}

__global__ __launch_bounds__(256) void fa_causal_wmma_lds_kernel(
    const float* __restrict__ Q, const float* __restrict__ K,
    const float* __restrict__ V, float* __restrict__ O) {
  __shared__ KVBuf sbuf[2];
  const int tid  = threadIdx.x;
  const int lane = tid & 31;
  const int wave = tid >> 5;
  const int bh   = blockIdx.x >> 4;
  const int wq0  = (blockIdx.x & 15) << 7;
  const int q0   = wq0 + wave * 16;
  const int n    = lane & 15;
  const int h    = lane >> 4;

  const size_t base = (size_t)bh * ATT_S * ATT_D;
  const float* __restrict__ Qb = Q + base;
  const float* __restrict__ Kb = K + base;
  const float* __restrict__ Vb = V + base;
  float* __restrict__       Ob = O + base;

  v16h qf[2];
  {
    const float* qrow = Qb + (size_t)(q0 + n) * ATT_D;
#pragma unroll
    for (int c = 0; c < 2; ++c) {
      load8_f16(qrow + 32 * c + 8 * h,      QSCALE, qf[c], 0);
      load8_f16(qrow + 32 * c + 16 + 8 * h, QSCALE, qf[c], 8);
    }
  }

  v8f acc[4];
#pragma unroll
  for (int c = 0; c < 4; ++c)
#pragma unroll
    for (int r = 0; r < 8; ++r) acc[c][r] = 0.0f;

  float m_run = -1e30f, l_run = 0.0f;
  const int nb = (wq0 >> 5) + 4;

  StageRegs sr = stage_load(Kb, Vb, 0, tid);
  stage_store(sbuf[0], sr, tid);
  __syncthreads();

  for (int blk = 0; blk < nb; ++blk) {
    const int  kb        = blk << 5;
    const bool have_next = (blk + 1 < nb);
    if (have_next) sr = stage_load(Kb, Vb, kb + 32, tid);

    if (kb < q0 + 16) {
      KVBuf& b = sbuf[blk & 1];
      v8f st[2];
#pragma unroll
      for (int t = 0; t < 2; ++t) {
        const int row = 16 * t + n;
        v16h ka0 = frag16(&b.ks[row][8 * h],      &b.ks[row][16 + 8 * h]);
        v16h ka1 = frag16(&b.ks[row][32 + 8 * h], &b.ks[row][48 + 8 * h]);
        v8f c0;
#pragma unroll
        for (int r = 0; r < 8; ++r) c0[r] = 0.0f;
        c0 = __builtin_amdgcn_wmma_f32_16x16x32_f16(false, ka0, false, qf[0],
                                                    (short)0, c0, false, false);
        c0 = __builtin_amdgcn_wmma_f32_16x16x32_f16(false, ka1, false, qf[1],
                                                    (short)0, c0, false, false);
        st[t] = c0;
      }
      if (kb + 31 > q0) {
        const int q = q0 + n;
#pragma unroll
        for (int t = 0; t < 2; ++t)
#pragma unroll
          for (int r = 0; r < 8; ++r)
            if (kb + 16 * t + 8 * h + r > q) st[t][r] = -1e30f;
      }
      float bm = -1e30f;
#pragma unroll
      for (int t = 0; t < 2; ++t)
#pragma unroll
        for (int r = 0; r < 8; ++r) bm = fmaxf(bm, st[t][r]);
      bm = fmaxf(bm, __shfl_xor(bm, 16, 32));
      const float m_new = fmaxf(m_run, bm);
      const float corr  = fast_exp2(m_run - m_new);
      v16h pb;
      float rsum = 0.0f;
#pragma unroll
      for (int t = 0; t < 2; ++t)
#pragma unroll
        for (int r = 0; r < 8; ++r) {
          const float p = fast_exp2(st[t][r] - m_new);
          rsum += p;
          pb[8 * t + r] = (_Float16)p;
        }
      rsum += __shfl_xor(rsum, 16, 32);
      l_run = l_run * corr + rsum;
      m_run = m_new;
#pragma unroll
      for (int c = 0; c < 4; ++c) {
#pragma unroll
        for (int r = 0; r < 8; ++r) acc[c][r] *= corr;
        const int vd = 16 * c + n;
        v16h va = frag16(&b.vt[vd][8 * h], &b.vt[vd][16 + 8 * h]);
        acc[c] = __builtin_amdgcn_wmma_f32_16x16x32_f16(false, va, false, pb,
                                                        (short)0, acc[c],
                                                        false, false);
      }
    }
    if (have_next) stage_store(sbuf[(blk + 1) & 1], sr, tid);
    __syncthreads();
  }

  const float inv_l = 1.0f / l_run;
  float* orow = Ob + (size_t)(q0 + n) * ATT_D;
#pragma unroll
  for (int c = 0; c < 4; ++c) {
    float4 o0, o1;
    o0.x = acc[c][0] * inv_l; o0.y = acc[c][1] * inv_l;
    o0.z = acc[c][2] * inv_l; o0.w = acc[c][3] * inv_l;
    o1.x = acc[c][4] * inv_l; o1.y = acc[c][5] * inv_l;
    o1.z = acc[c][6] * inv_l; o1.w = acc[c][7] * inv_l;
    *(float4*)(orow + 16 * c + 8 * h)     = o0;
    *(float4*)(orow + 16 * c + 8 * h + 4) = o1;
  }
}

// ---------------------------------------------------------------------------
extern "C" void kernel_launch(void* const* d_in, const int* in_sizes, int n_in,
                              void* d_out, int out_size, void* d_ws, size_t ws_size,
                              hipStream_t stream) {
  (void)in_sizes; (void)n_in; (void)out_size;
  const float* Q = (const float*)d_in[0];
  const float* K = (const float*)d_in[1];
  const float* V = (const float*)d_in[2];
  // d_in[3] (causal mask) is deterministic tril; computed analytically.
  float* Out = (float*)d_out;

  if (ws_size >= WS_NEED) {
    _Float16* Qh  = (_Float16*)d_ws;
    _Float16* Kh  = Qh + ARR_ELT;
    _Float16* Vth = Kh + ARR_ELT;
    // Pre-pass: one-time f32->f16 (+scale, +V transpose) into workspace.
    hipLaunchKernelGGL(cvt_qk_kernel, dim3(2 * ARR_ELT / 8 / 256), dim3(256),
                       0, stream, Q, K, Qh, Kh);
    hipLaunchKernelGGL(transpose_v_kernel, dim3(ATT_BH * 32), dim3(256),
                       0, stream, V, Vth);
    hipLaunchKernelGGL(fa_causal_wmma_async_kernel, dim3(1024), dim3(256),
                       0, stream, Qh, Kh, Vth, Out);
  } else {
    hipLaunchKernelGGL(fa_causal_wmma_lds_kernel, dim3(1024), dim3(256),
                       0, stream, Q, K, V, Out);
  }
}
